// MASA_54022098649572
// MI455X (gfx1250) — matrix operand order repaired and checked by reading, not compiled, not run on hardware
//
#include <hip/hip_runtime.h>
#include <hip/hip_bf16.h>
#include <math.h>

typedef _Float16 f16;
typedef _Float16 v8h  __attribute__((ext_vector_type(8)));
typedef _Float16 v16h __attribute__((ext_vector_type(16)));
typedef float    v8f  __attribute__((ext_vector_type(8)));

#define BATCH 4
#define CIN   192
#define OC    768
#define HEADS 8
#define CH    24
#define CP    32
#define WIDTH 128
#define HW    16384
#define NAG   64

static __device__ inline v16h cat16(v8h lo, v8h hi) {
  v16h r;
#pragma unroll
  for (int i = 0; i < 8; ++i) { r[i] = lo[i]; r[i + 8] = hi[i]; }
  return r;
}
static __device__ inline v8f v8f_zero() {
  v8f r;
#pragma unroll
  for (int i = 0; i < 8; ++i) r[i] = 0.f;
  return r;
}
#define WMMA16(A, B, C) __builtin_amdgcn_wmma_f32_16x16x32_f16(false, (A), false, (B), (short)0, (C), false, false)

static __device__ inline void redmax16(v8f& v) {
#pragma unroll
  for (int m = 1; m < 16; m <<= 1) {
#pragma unroll
    for (int i = 0; i < 8; ++i) {
      float o = __shfl_xor(v[i], m);
      v[i] = fmaxf(v[i], o);
    }
  }
}
static __device__ inline void redsum16(v8f& v) {
#pragma unroll
  for (int m = 1; m < 16; m <<= 1) {
#pragma unroll
    for (int i = 0; i < 8; ++i) {
      float o = __shfl_xor(v[i], m);
      v[i] += o;
    }
  }
}

// ---------------- K0: pack w_qkv to f16 ----------------
__global__ void k_pack_w(const float* __restrict__ wq, f16* __restrict__ wh, int n) {
  int i = blockIdx.x * 256 + threadIdx.x;
  if (i < n) wh[i] = (f16)wq[i];
}

// ---------------- K1: conv1x1 as WMMA GEMM  D[oc,n] = W[oc,ic] * X[ic,n] ----
__global__ __launch_bounds__(256) void k_conv1x1(const float* __restrict__ x,
                                                const f16* __restrict__ wh,
                                                f16* __restrict__ pre) {
  const int b = blockIdx.z;
  const int m_t = blockIdx.y * 128;  // oc tile
  const int n_t = blockIdx.x * 128;  // pixel tile
  const int tid = threadIdx.x, wv = tid >> 5, lane = tid & 31;
  const int lo16 = (lane < 16), l15 = lane & 15;

  __shared__ f16 As[128][CIN];  // whole W tile [oc][k], staged once
  __shared__ f16 Bs[128][32];   // X K-slab, transposed to [n][k]

  // ---- stage entire W tile: pure linear global->LDS copy ----
  {
    const f16* gsrc = wh + (size_t)m_t * CIN;
#if __has_builtin(__builtin_amdgcn_global_load_async_to_lds_b128)
    typedef int vsi4 __attribute__((vector_size(16)));
    typedef __attribute__((address_space(1))) vsi4* gp_t;
    typedef __attribute__((address_space(3))) vsi4* lp_t;
#pragma unroll
    for (int i = 0; i < 6; ++i) {
      int off = (tid + i * 256) * 16;  // halves
      __builtin_amdgcn_global_load_async_to_lds_b128(
          (gp_t)(gsrc + off), (lp_t)(&As[0][0] + off), 0, 0);
    }
#if __has_builtin(__builtin_amdgcn_s_wait_asynccnt)
    __builtin_amdgcn_s_wait_asynccnt(0);
#else
    asm volatile("s_wait_asynccnt 0" ::: "memory");
#endif
#else
#pragma unroll
    for (int i = 0; i < 6; ++i) {
      int off = (tid + i * 256) * 16;
      const v8h* s = (const v8h*)(gsrc + off);
      v8h* d = (v8h*)(&As[0][0] + off);
      d[0] = s[0];
      d[1] = s[1];
    }
#endif
  }

  v8f acc[8];
#pragma unroll
  for (int j = 0; j < 8; ++j) acc[j] = v8f_zero();

  const int m0 = wv * 16;
  for (int k0 = 0; k0 < CIN; k0 += 32) {
    __syncthreads();
    {  // stage X slab: f32 -> packed f16x2, transposed to [n][k]
      int kp = tid & 15;        // k pair: k = 2*kp
      int ng = (tid >> 4) * 8;  // 8 pixels
      const float* x0 = x + (size_t)(b * CIN + k0 + 2 * kp) * HW + n_t + ng;
      const float* x1 = x0 + HW;
      if (k0 + 32 < CIN) __builtin_prefetch(x0 + 32 * HW, 0, 1);
#pragma unroll
      for (int j = 0; j < 8; ++j) {
        union { f16 h[2]; unsigned u; } pk;
        pk.h[0] = (f16)x0[j];
        pk.h[1] = (f16)x1[j];
        *(unsigned*)(&Bs[ng + j][2 * kp]) = pk.u;
      }
    }
    __syncthreads();

    v8h alo = *(const v8h*)(&As[m0 + l15][k0 + (lo16 ? 0 : 16)]);
    v8h ahi = *(const v8h*)(&As[m0 + l15][k0 + (lo16 ? 0 : 16) + 8]);
    v16h Am = cat16(alo, ahi);
    v16h Bm[8];
#pragma unroll
    for (int j = 0; j < 8; ++j) {
      v8h blo = *(const v8h*)(&Bs[j * 16 + l15][lo16 ? 0 : 16]);
      v8h bhi = *(const v8h*)(&Bs[j * 16 + l15][(lo16 ? 0 : 16) + 8]);
      Bm[j] = cat16(blo, bhi);
    }
#pragma unroll
    for (int j = 0; j < 8; ++j) acc[j] = WMMA16(Am, Bm[j], acc[j]);
  }
  const int mg = m_t + m0;
  const int hi8 = lo16 ? 0 : 8;
#pragma unroll
  for (int j = 0; j < 8; ++j) {
#pragma unroll
    for (int r = 0; r < 8; ++r) {
      int oc = mg + r + hi8;
      int n = n_t + j * 16 + l15;
      pre[(size_t)(b * OC + oc) * HW + n] = (f16)acc[j][r];
    }
  }
}

// ---------------- K2: dwconv3x3 + l2norm(q,k) + head transpose (pad 24->32) --
__global__ __launch_bounds__(256) void k_dwnorm(const f16* __restrict__ pre,
                                                const float* __restrict__ wdw,
                                                f16* __restrict__ Qn,
                                                f16* __restrict__ Kn,
                                                f16* __restrict__ Vh) {
  const int bh = blockIdx.y;
  const int b = bh >> 3, h = bh & 7;
  const int tid = threadIdx.x;
  const int px = blockIdx.x * 256 + tid;
  const int y = px >> 7, xx0 = px & 127;

  __shared__ float wl[3 * 24 * 9];
  for (int i = tid; i < 3 * 24 * 9; i += 256) {
    int part = i / 216, rem = i % 216, cc = rem / 9, k = rem % 9;
    wl[i] = wdw[(part * CIN + h * CH + cc) * 9 + k];
  }
  __syncthreads();

  float out3[3][CH];
#pragma unroll
  for (int part = 0; part < 3; ++part) {
#pragma unroll
    for (int cc = 0; cc < CH; ++cc) {
      const f16* src = pre + (size_t)(b * OC + part * CIN + h * CH + cc) * HW;
      float a = 0.f;
#pragma unroll
      for (int dy = -1; dy <= 1; ++dy) {
        int yy = y + dy;
        if (yy < 0 || yy >= WIDTH) continue;
#pragma unroll
        for (int dx = -1; dx <= 1; ++dx) {
          int xc = xx0 + dx;
          if (xc < 0 || xc >= WIDTH) continue;
          a += (float)src[yy * WIDTH + xc] * wl[(part * CH + cc) * 9 + (dy + 1) * 3 + (dx + 1)];
        }
      }
      out3[part][cc] = a;
    }
  }
  float inv[2];
#pragma unroll
  for (int part = 0; part < 2; ++part) {
    float s = 0.f;
#pragma unroll
    for (int cc = 0; cc < CH; ++cc) s += out3[part][cc] * out3[part][cc];
    inv[part] = 1.f / fmaxf(sqrtf(s), 1e-12f);
  }
  f16* dsts[3] = {Qn, Kn, Vh};
#pragma unroll
  for (int part = 0; part < 3; ++part) {
    v8h buf4[4];
    f16* bp = (f16*)buf4;
    float sc = (part < 2) ? inv[part] : 1.f;
#pragma unroll
    for (int cc = 0; cc < CH; ++cc) bp[cc] = (f16)(out3[part][cc] * sc);
#pragma unroll
    for (int cc = CH; cc < CP; ++cc) bp[cc] = (f16)0.f;
    v8h* dst = (v8h*)(dsts[part] + (size_t)(bh * HW + px) * CP);
#pragma unroll
    for (int j = 0; j < 4; ++j) dst[j] = buf4[j];
  }
}

// ---------------- K3: agent pooling (dwconv of 'a' + 16x16 mean pool) -------
__global__ __launch_bounds__(256) void k_agentpool(const f16* __restrict__ pre,
                                                   const float* __restrict__ wdw,
                                                   f16* __restrict__ agent) {
  const int c = blockIdx.x;  // 0..191
  const int b = blockIdx.y;
  const int ch = 3 * CIN + c;
  const int tid = threadIdx.x;
  __shared__ float w9[9];
  __shared__ float partial[256];
  if (tid < 9) w9[tid] = wdw[ch * 9 + tid];
  __syncthreads();

  const int m = tid >> 2, part = tid & 3;
  const int cy = m >> 3, cx = m & 7;
  const f16* src = pre + (size_t)(b * OC + ch) * HW;
  float s = 0.f;
  for (int r = 0; r < 4; ++r) {
    int y = cy * 16 + part * 4 + r;
    for (int col = 0; col < 16; ++col) {
      int x = cx * 16 + col;
      float a = 0.f;
#pragma unroll
      for (int dy = -1; dy <= 1; ++dy) {
        int yy = y + dy;
        if (yy < 0 || yy >= WIDTH) continue;
#pragma unroll
        for (int dx = -1; dx <= 1; ++dx) {
          int xc = x + dx;
          if (xc < 0 || xc >= WIDTH) continue;
          a += (float)src[yy * WIDTH + xc] * w9[(dy + 1) * 3 + (dx + 1)];
        }
      }
      s += a;
    }
  }
  partial[tid] = s;
  __syncthreads();
  if (part == 0) {
    float t = partial[tid] + partial[tid + 1] + partial[tid + 2] + partial[tid + 3];
    agent[((size_t)(b * HEADS + c / CH) * NAG + m) * CP + (c % CH)] = (f16)(t * (1.f / 256.f));
  }
}

// ---------------- K4: k_attn softmax + agent_v  (per (b,h) workgroup) -------
__global__ __launch_bounds__(256) void k_agentv(const f16* __restrict__ Kn,
                                                const f16* __restrict__ Vh,
                                                const f16* __restrict__ agent,
                                                const float* __restrict__ temp,
                                                f16* __restrict__ agentv) {
  const int bh = blockIdx.x;
  const int h = bh & 7;
  const int tid = threadIdx.x, w = tid >> 5, lane = tid & 31;
  const int lo16 = (lane < 16), l15 = lane & 15;
  const int hi8 = lo16 ? 0 : 8;
  const float T = temp[h];

  __shared__ float AVacc[NAG][CP];
  __shared__ float slots[NAG][8];
  __shared__ float rowmaxL[NAG];
  __shared__ float rowsumL[NAG];
  __shared__ f16 Pls[8][NAG][32];
  __shared__ f16 Vls[8][32][32];

  for (int i = tid; i < NAG * CP; i += 256) ((float*)AVacc)[i] = 0.f;
  __syncthreads();

  v16h Aag[4];
#pragma unroll
  for (int s = 0; s < 4; ++s) {
    const f16* ap = agent + (size_t)(bh * NAG + s * 16 + l15) * CP;
    v8h lo = *(const v8h*)(ap + (lo16 ? 0 : 16));
    v8h hi = *(const v8h*)(ap + (lo16 ? 0 : 16) + 8);
    Aag[s] = cat16(lo, hi);
  }

  // Pass 1: row max over n
  v8f rmax[4];
#pragma unroll
  for (int s = 0; s < 4; ++s)
#pragma unroll
    for (int e = 0; e < 8; ++e) rmax[s][e] = -1e30f;

  for (int i = 0; i < 128; ++i) {
    int n0 = (i * 8 + w) * 16;
    const f16* kp = Kn + (size_t)(bh * HW + n0 + l15) * CP + (lo16 ? 0 : 16);
    if (i + 1 < 128) __builtin_prefetch(kp + 128 * CP, 0, 1);
    v16h Bk = cat16(*(const v8h*)kp, *(const v8h*)(kp + 8));
#pragma unroll
    for (int s = 0; s < 4; ++s) {
      v8f S = WMMA16(Aag[s], Bk, v8f_zero());
#pragma unroll
      for (int e = 0; e < 8; ++e) rmax[s][e] = fmaxf(rmax[s][e], S[e] * T);
    }
  }
#pragma unroll
  for (int s = 0; s < 4; ++s) redmax16(rmax[s]);
  if (l15 == 0) {
#pragma unroll
    for (int s = 0; s < 4; ++s)
#pragma unroll
      for (int r = 0; r < 8; ++r) slots[s * 16 + r + hi8][w] = rmax[s][r];
  }
  __syncthreads();
  if (tid < NAG) {
    float mx = -1e30f;
    for (int j = 0; j < 8; ++j) mx = fmaxf(mx, slots[tid][j]);
    rowmaxL[tid] = mx;
  }
  __syncthreads();

  v8f rm[4];
#pragma unroll
  for (int s = 0; s < 4; ++s)
#pragma unroll
    for (int r = 0; r < 8; ++r) rm[s][r] = rowmaxL[s * 16 + r + hi8];

  // Pass 2: unnormalized exp, accumulate row sums and AV = P * V
  v8f sacc[4];
  v8f AV[4][2];
#pragma unroll
  for (int s = 0; s < 4; ++s) {
    sacc[s] = v8f_zero();
    AV[s][0] = v8f_zero();
    AV[s][1] = v8f_zero();
  }

  for (int i = 0; i < 64; ++i) {
    int n0 = (i * 8 + w) * 32;
#pragma unroll
    for (int h2 = 0; h2 < 2; ++h2) {
      int nb = n0 + h2 * 16;
      const f16* kp = Kn + (size_t)(bh * HW + nb + l15) * CP + (lo16 ? 0 : 16);
      if (i + 1 < 64) __builtin_prefetch(kp + 256 * CP, 0, 1);
      v16h Bk = cat16(*(const v8h*)kp, *(const v8h*)(kp + 8));
#pragma unroll
      for (int s = 0; s < 4; ++s) {
        v8f S = WMMA16(Aag[s], Bk, v8f_zero());
#pragma unroll
        for (int e = 0; e < 8; ++e) {
          float p = __expf(S[e] * T - rm[s][e]);
          sacc[s][e] += p;
          Pls[w][s * 16 + e + hi8][h2 * 16 + l15] = (f16)p;
        }
      }
    }
    {  // stage V chunk transposed: Vls[c][n_local]
      int n = n0 + lane;
      const f16* vp = Vh + (size_t)(bh * HW + n) * CP;
      if (i + 1 < 64) __builtin_prefetch(vp + 256 * CP, 0, 1);
      v8h v0 = *(const v8h*)(vp), v1 = *(const v8h*)(vp + 8);
      v8h v2 = *(const v8h*)(vp + 16), v3 = *(const v8h*)(vp + 24);
#pragma unroll
      for (int cc = 0; cc < 8; ++cc) {
        Vls[w][cc][lane] = v0[cc];
        Vls[w][8 + cc][lane] = v1[cc];
        Vls[w][16 + cc][lane] = v2[cc];
        Vls[w][24 + cc][lane] = v3[cc];
      }
    }
    asm volatile("s_wait_dscnt 0" ::: "memory");
    // preload all operands, then back-to-back WMMA chain
    v16h PaV[4];
#pragma unroll
    for (int s = 0; s < 4; ++s) {
      v8h plo = *(const v8h*)(&Pls[w][s * 16 + l15][lo16 ? 0 : 16]);
      v8h phi = *(const v8h*)(&Pls[w][s * 16 + l15][(lo16 ? 0 : 16) + 8]);
      PaV[s] = cat16(plo, phi);
    }
    v16h VbV[2];
#pragma unroll
    for (int t = 0; t < 2; ++t) {
      int c = t * 16 + l15;
      v8h blo = *(const v8h*)(&Vls[w][c][lo16 ? 0 : 16]);
      v8h bhi = *(const v8h*)(&Vls[w][c][(lo16 ? 0 : 16) + 8]);
      VbV[t] = cat16(blo, bhi);
    }
#pragma unroll
    for (int s = 0; s < 4; ++s)
#pragma unroll
      for (int t = 0; t < 2; ++t) AV[s][t] = WMMA16(PaV[s], VbV[t], AV[s][t]);
  }
#pragma unroll
  for (int s = 0; s < 4; ++s) redsum16(sacc[s]);
  if (l15 == 0) {
#pragma unroll
    for (int s = 0; s < 4; ++s)
#pragma unroll
      for (int r = 0; r < 8; ++r) slots[s * 16 + r + hi8][w] = sacc[s][r];
  }
#pragma unroll
  for (int s = 0; s < 4; ++s)
#pragma unroll
    for (int t = 0; t < 2; ++t)
#pragma unroll
      for (int r = 0; r < 8; ++r)
        atomicAdd(&AVacc[s * 16 + r + hi8][t * 16 + l15], AV[s][t][r]);
  __syncthreads();
  if (tid < NAG) {
    float ss = 0.f;
    for (int j = 0; j < 8; ++j) ss += slots[tid][j];
    rowsumL[tid] = ss;
  }
  __syncthreads();
  for (int e = tid; e < NAG * CP; e += 256) {
    int m = e >> 5, c = e & 31;
    agentv[(size_t)(bh * NAG + m) * CP + c] = (f16)(AVacc[m][c] / rowsumL[m]);
  }
}

// ---------------- K5: q_attn softmax + out = softmax(Q*agentT)*agent_v ------
__global__ __launch_bounds__(256) void k_out(const f16* __restrict__ Qn,
                                             const f16* __restrict__ agent,
                                             const f16* __restrict__ agentv,
                                             const float* __restrict__ temp,
                                             float* __restrict__ outbuf) {
  const int bh = blockIdx.y;
  const int b = bh >> 3, h = bh & 7;
  const int tid = threadIdx.x, w = tid >> 5, lane = tid & 31;
  const int lo16 = (lane < 16), l15 = lane & 15;
  const int hi8 = lo16 ? 0 : 8;
  const float T = temp[h];

  __shared__ f16 avT[CP][NAG];   // agent_v transposed
  __shared__ f16 Pw[8][16][NAG]; // per-wave probability tile

  for (int e = tid; e < CP * NAG; e += 256) {
    int m = e & 63, c = e >> 6;
    avT[c][m] = agentv[(size_t)(bh * NAG + m) * CP + c];
  }
  __syncthreads();

  v16h Bag[4];
#pragma unroll
  for (int s = 0; s < 4; ++s) {
    const f16* ap = agent + (size_t)(bh * NAG + s * 16 + l15) * CP + (lo16 ? 0 : 16);
    Bag[s] = cat16(*(const v8h*)ap, *(const v8h*)(ap + 8));
  }
  const int n0 = blockIdx.x * 128 + w * 16;
  const f16* qp = Qn + (size_t)(bh * HW + n0 + l15) * CP;
  v16h Qa = cat16(*(const v8h*)(qp + (lo16 ? 0 : 16)),
                  *(const v8h*)(qp + (lo16 ? 0 : 16) + 8));

  v8f S[4];
#pragma unroll
  for (int s = 0; s < 4; ++s) {
    S[s] = WMMA16(Qa, Bag[s], v8f_zero());
#pragma unroll
    for (int e = 0; e < 8; ++e) S[s][e] *= T;
  }
  v8f rmax = S[0];
#pragma unroll
  for (int s = 1; s < 4; ++s)
#pragma unroll
    for (int e = 0; e < 8; ++e) rmax[e] = fmaxf(rmax[e], S[s][e]);
  redmax16(rmax);
  v8f rsum = v8f_zero();
#pragma unroll
  for (int s = 0; s < 4; ++s)
#pragma unroll
    for (int e = 0; e < 8; ++e) {
      float p = __expf(S[s][e] - rmax[e]);
      S[s][e] = p;
      rsum[e] += p;
    }
  redsum16(rsum);

#pragma unroll
  for (int s = 0; s < 4; ++s)
#pragma unroll
    for (int r = 0; r < 8; ++r) Pw[w][r + hi8][s * 16 + l15] = (f16)S[s][r];
  asm volatile("s_wait_dscnt 0" ::: "memory");

  // preload both K-chunks of P and agent_v^T, then WMMA chain
  v16h Pa2[2];
  v16h avB[2][2];
#pragma unroll
  for (int kcI = 0; kcI < 2; ++kcI) {
    int kc = kcI * 32;
    v8h plo = *(const v8h*)(&Pw[w][l15][kc + (lo16 ? 0 : 16)]);
    v8h phi = *(const v8h*)(&Pw[w][l15][kc + (lo16 ? 0 : 16) + 8]);
    Pa2[kcI] = cat16(plo, phi);
#pragma unroll
    for (int t = 0; t < 2; ++t) {
      int c = t * 16 + l15;
      v8h blo = *(const v8h*)(&avT[c][kc + (lo16 ? 0 : 16)]);
      v8h bhi = *(const v8h*)(&avT[c][kc + (lo16 ? 0 : 16) + 8]);
      avB[kcI][t] = cat16(blo, bhi);
    }
  }
  v8f D[2] = {v8f_zero(), v8f_zero()};
#pragma unroll
  for (int kcI = 0; kcI < 2; ++kcI)
#pragma unroll
    for (int t = 0; t < 2; ++t) D[t] = WMMA16(Pa2[kcI], avB[kcI][t], D[t]);

#pragma unroll
  for (int t = 0; t < 2; ++t) {
    int c = t * 16 + l15;
    if (c < CH) {
#pragma unroll
      for (int r = 0; r < 8; ++r) {
        int px = n0 + r + hi8;
        outbuf[(size_t)(b * CIN + h * CH + c) * HW + px] = D[t][r] / rsum[r];
      }
    }
  }
}

// ---------------- K6: simam (per-channel mean/var + sigmoid gate) -----------
__global__ __launch_bounds__(256) void k_simam(const float* __restrict__ outbuf,
                                               float* __restrict__ dout) {
  const int bc = blockIdx.x;
  const int tid = threadIdx.x;
  const float* src = outbuf + (size_t)bc * HW;
  float v[64];
  float s = 0.f;
#pragma unroll
  for (int i = 0; i < 64; ++i) {
    v[i] = src[tid + i * 256];
    s += v[i];
  }
  __shared__ float red[256];
  red[tid] = s;
  __syncthreads();
  for (int o = 128; o > 0; o >>= 1) {
    if (tid < o) red[tid] += red[tid + o];
    __syncthreads();
  }
  float mu = red[0] * (1.f / 16384.f);
  __syncthreads();
  float s2 = 0.f;
#pragma unroll
  for (int i = 0; i < 64; ++i) {
    float d = v[i] - mu;
    s2 += d * d;
  }
  red[tid] = s2;
  __syncthreads();
  for (int o = 128; o > 0; o >>= 1) {
    if (tid < o) red[tid] += red[tid + o];
    __syncthreads();
  }
  float denom = 4.f * (red[0] * (1.f / 16383.f) + 1e-4f);
  float* dst = dout + (size_t)bc * HW;
#pragma unroll
  for (int i = 0; i < 64; ++i) {
    float d = v[i] - mu;
    float y = d * d / denom + 0.5f;
    dst[tid + i * 256] = v[i] * (1.f / (1.f + __expf(-y)));
  }
}

// ---------------- workspace layout (bytes) ----------------------------------
#define OFF_WH     ((size_t)0)
#define OFF_PRE    ((size_t)294912)
#define OFF_QN     (OFF_PRE + (size_t)BATCH * OC * HW * 2)
#define OFF_KN     (OFF_QN + (size_t)BATCH * HEADS * HW * CP * 2)
#define OFF_V      (OFF_KN + (size_t)BATCH * HEADS * HW * CP * 2)
#define OFF_AGENT  (OFF_V + (size_t)BATCH * HEADS * HW * CP * 2)
#define OFF_AGENTV (OFF_AGENT + (size_t)BATCH * HEADS * NAG * CP * 2)
#define OFF_OUT    (OFF_AGENTV + (size_t)BATCH * HEADS * NAG * CP * 2)

extern "C" void kernel_launch(void* const* d_in, const int* in_sizes, int n_in,
                              void* d_out, int out_size, void* d_ws, size_t ws_size,
                              hipStream_t stream) {
  const float* x     = (const float*)d_in[0];
  const float* w_qkv = (const float*)d_in[1];
  const float* w_dw  = (const float*)d_in[2];
  const float* temp  = (const float*)d_in[3];
  float* dout = (float*)d_out;

  char* ws = (char*)d_ws;
  f16* wh     = (f16*)(ws + OFF_WH);
  f16* pre    = (f16*)(ws + OFF_PRE);
  f16* Qn     = (f16*)(ws + OFF_QN);
  f16* Kn     = (f16*)(ws + OFF_KN);
  f16* Vh     = (f16*)(ws + OFF_V);
  f16* agent  = (f16*)(ws + OFF_AGENT);
  f16* agentv = (f16*)(ws + OFF_AGENTV);
  float* outb = (float*)(ws + OFF_OUT);

  k_pack_w<<<(OC * CIN + 255) / 256, 256, 0, stream>>>(w_qkv, wh, OC * CIN);
  k_conv1x1<<<dim3(HW / 128, OC / 128, BATCH), 256, 0, stream>>>(x, wh, pre);
  (void)hipMemsetAsync(agent, 0, (size_t)BATCH * HEADS * NAG * CP * 2, stream);
  k_dwnorm<<<dim3(HW / 256, BATCH * HEADS), 256, 0, stream>>>(pre, w_dw, Qn, Kn, Vh);
  k_agentpool<<<dim3(CIN, BATCH), 256, 0, stream>>>(pre, w_dw, agent);
  k_agentv<<<BATCH * HEADS, 256, 0, stream>>>(Kn, Vh, agent, temp, agentv);
  k_out<<<dim3(HW / 128, BATCH * HEADS), 256, 0, stream>>>(Qn, agent, agentv, temp, outb);
  k_simam<<<BATCH * CIN, 256, 0, stream>>>(outb, dout);
}
